// PureSTGCN_83580063580899
// MI455X (gfx1250) — matrix operand order repaired and checked by reading, not compiled, not run on hardware
//
#include <hip/hip_runtime.h>
#include <math.h>

// ---------------------------------------------------------------------------
// PureSTGCN on gfx1250 (MI455X). Dense GEMMs via v_wmma_f32_16x16x32_f16,
// f16 activations, LDS-staged weight tiles, 4 WMMAs per A-fragment with
// all B fragments loaded up-front per K-step (staggered ds waits).
// ---------------------------------------------------------------------------

typedef __attribute__((ext_vector_type(16))) _Float16 v16h;
typedef __attribute__((ext_vector_type(8)))  _Float16 v8h;
typedef __attribute__((ext_vector_type(8)))  float    v8f;

#define NBATCH 32
#define TLEN   50
#define NJOINT 55
#define NNODES (NBATCH * TLEN * NJOINT)   // 88000, divisible by 16
#define NCLS   100

__device__ __forceinline__ float gelu_exact(float x) {
    return 0.5f * x * (1.0f + erff(x * 0.70710678118654752440f));
}

__device__ __forceinline__ v16h join16(v8h lo, v8h hi) {
    union { v16h v; v8h h[2]; } u;
    u.h[0] = lo; u.h[1] = hi;
    return u.v;
}

// ------------------------- WMMA GEMM: out = Xh @ Wh^T (+bias) --------------
// Xh: [M,K] f16 row-major, Wh: [N,K] f16 row-major, out: [M,N] f32.
// Block = 8 waves; each wave computes a 16x64 output strip (4 WMMA acc).
// Weight tile-column (64 rows x K) staged in LDS once per block.
__global__ __launch_bounds__(256) void k_wmma_gemm(
    const _Float16* __restrict__ Xh, const _Float16* __restrict__ Wh,
    const float* __restrict__ bias, float* __restrict__ out,
    int M, int N, int K)
{
    __shared__ _Float16 sW[64 * 256];            // up to 32 KB (K <= 256)

    const int tid  = threadIdx.x;
    const int wave = tid >> 5;
    const int lane = tid & 31;
    const int n0   = blockIdx.y * 64;            // 64-col N group

    // cooperative load of W rows n0..n0+63 (K f16 each) into LDS, 16B chunks
    for (int idx = tid * 8; idx < 64 * K; idx += 256 * 8) {
        int r = idx / K, k = idx - r * K;        // K%8==0 -> chunk stays in-row
        *(v8h*)(&sW[idx]) = *(const v8h*)(&Wh[(size_t)(n0 + r) * K + k]);
    }
    __syncthreads();

    const int tileM = blockIdx.x * 8 + wave;
    if (tileM * 16 < M) {                        // wave-uniform: EXEC stays full
        const int l15 = lane & 15;
        const int hi  = lane >> 4;               // 0: lanes 0-15, 1: lanes 16-31
        const int kbA = hi * 16;                 // A K-base (contiguous K, matches B)
        const int kbB = hi * 16;                 // B K-base (contiguous K = 16*hi + s, matches A)
        const _Float16* __restrict__ xrow = Xh + (size_t)(tileM * 16 + l15) * K;

        v8f acc0 = {}, acc1 = {}, acc2 = {}, acc3 = {};
        for (int k0 = 0; k0 < K; k0 += 32) {
            const _Float16* xa = xrow + k0 + kbA;
            v16h a = join16(*(const v8h*)xa, *(const v8h*)(xa + 8));

            // issue all 4 B-fragment loads (8x ds_load_b128) before any WMMA
            const _Float16* sb = &sW[(size_t)l15 * K + k0 + kbB];
            const _Float16* bp0 = sb;
            const _Float16* bp1 = sb + (size_t)16 * K;
            const _Float16* bp2 = sb + (size_t)32 * K;
            const _Float16* bp3 = sb + (size_t)48 * K;
            v16h b0 = join16(*(const v8h*)bp0, *(const v8h*)(bp0 + 8));
            v16h b1 = join16(*(const v8h*)bp1, *(const v8h*)(bp1 + 8));
            v16h b2 = join16(*(const v8h*)bp2, *(const v8h*)(bp2 + 8));
            v16h b3 = join16(*(const v8h*)bp3, *(const v8h*)(bp3 + 8));

            acc0 = __builtin_amdgcn_wmma_f32_16x16x32_f16(false, a, false, b0, (short)0, acc0, false, false);
            acc1 = __builtin_amdgcn_wmma_f32_16x16x32_f16(false, a, false, b1, (short)0, acc1, false, false);
            acc2 = __builtin_amdgcn_wmma_f32_16x16x32_f16(false, a, false, b2, (short)0, acc2, false, false);
            acc3 = __builtin_amdgcn_wmma_f32_16x16x32_f16(false, a, false, b3, (short)0, acc3, false, false);
        }

        const int rbase = tileM * 16 + hi * 8;   // C/D: VGPR r -> M = r + 8*hi
#pragma unroll
        for (int j = 0; j < 4; ++j) {
            const v8f& acc = (j == 0) ? acc0 : (j == 1) ? acc1 : (j == 2) ? acc2 : acc3;
            const int col = n0 + j * 16 + l15;
            const float bval = bias ? bias[col] : 0.0f;
#pragma unroll
            for (int r = 0; r < 8; ++r)
                out[(size_t)(rbase + r) * N + col] = acc[r] + bval;
        }
    }
}

// ------------------------- small helper kernels ----------------------------
__global__ void k_fill(float* __restrict__ p, int n, float v) {
    int i = blockIdx.x * blockDim.x + threadIdx.x;
    if (i < n) p[i] = v;
}

__global__ void k_cvt_f32_f16(const float* __restrict__ src, _Float16* __restrict__ dst, int n) {
    int i = blockIdx.x * blockDim.x + threadIdx.x;
    if (i < n) dst[i] = (_Float16)src[i];
}

__global__ void k_deg_count(const int* __restrict__ dst, float* __restrict__ deg, int E) {
    int e = blockIdx.x * blockDim.x + threadIdx.x;
    if (e < E) atomicAdd(&deg[dst[e]], 1.0f);
}

__global__ void k_rsqrt_inplace(float* __restrict__ p, int n) {
    int i = blockIdx.x * blockDim.x + threadIdx.x;
    if (i < n) p[i] = rsqrtf(p[i]);
}

// input projection: [M,6] f32 -> [M,64] f16
__global__ void k_inproj(const float* __restrict__ x, const float* __restrict__ w,
                         const float* __restrict__ b, _Float16* __restrict__ out, int M) {
    int i = blockIdx.x * blockDim.x + threadIdx.x;   // over M*64
    if (i >= M * 64) return;
    int node = i >> 6, o = i & 63;
    const float* xp = x + node * 6;
    float s = b[o];
#pragma unroll
    for (int k = 0; k < 6; ++k) s = fmaf(w[o * 6 + k], xp[k], s);
    out[i] = (_Float16)s;
}

// agg[i] = h[i]*dinv[node]^2 + bias[c]   (self-loop message + GCN bias)
__global__ void k_agg_init(const float* __restrict__ h, const float* __restrict__ dinv,
                           const float* __restrict__ bias, float* __restrict__ agg,
                           int total, int C) {
    int i = blockIdx.x * blockDim.x + threadIdx.x;
    if (i >= total) return;
    int node = i / C, c = i - node * C;
    float d = dinv[node];
    agg[i] = h[i] * d * d + bias[c];
}

// scatter: agg[dst] += h[src] * dinv[src]*dinv[dst], thread per (edge, 4-ch group)
__global__ void k_agg_edges(const float* __restrict__ h, const int* __restrict__ src,
                            const int* __restrict__ dst, const float* __restrict__ dinv,
                            float* __restrict__ agg, int E, int C) {
    const int g4 = C >> 2;
    long long i = (long long)blockIdx.x * blockDim.x + threadIdx.x;
    if (i >= (long long)E * g4) return;
    int e  = (int)(i / g4);
    int c4 = (int)(i - (long long)e * g4) * 4;
    int s = src[e], d = dst[e];
    float nrm = dinv[s] * dinv[d];
    const float4 m = *(const float4*)(h + (size_t)s * C + c4);
    float* ap = agg + (size_t)d * C + c4;
    atomicAdd(ap + 0, m.x * nrm);
    atomicAdd(ap + 1, m.y * nrm);
    atomicAdd(ap + 2, m.z * nrm);
    atomicAdd(ap + 3, m.w * nrm);
}

// out_f16 = gelu(batchnorm(agg) + res_f32)   (projected residual path)
__global__ void k_bn_resf_gelu(const float* __restrict__ agg, const float* __restrict__ rm,
                               const float* __restrict__ rv, const float* __restrict__ g,
                               const float* __restrict__ bb, const float* __restrict__ res,
                               _Float16* __restrict__ out, int total, int C) {
    int i = blockIdx.x * blockDim.x + threadIdx.x;
    if (i >= total) return;
    int c = i % C;
    float hn = (agg[i] - rm[c]) * rsqrtf(rv[c] + 1e-5f) * g[c] + bb[c];
    out[i] = (_Float16)gelu_exact(hn + res[i]);
}

// out_f16 = gelu(batchnorm(agg) + res_f16)   (identity residual path; in-place ok)
__global__ void k_bn_resh_gelu(const float* __restrict__ agg, const float* __restrict__ rm,
                               const float* __restrict__ rv, const float* __restrict__ g,
                               const float* __restrict__ bb, const _Float16* __restrict__ res,
                               _Float16* __restrict__ out, int total, int C) {
    int i = blockIdx.x * blockDim.x + threadIdx.x;
    if (i >= total) return;
    int c = i % C;
    float hn = (agg[i] - rm[c]) * rsqrtf(rv[c] + 1e-5f) * g[c] + bb[c];
    out[i] = (_Float16)gelu_exact(hn + (float)res[i]);
}

// mean over joints: [B,T,N,256] f16 -> [B,T,256] f32
__global__ void k_mean_joints(const _Float16* __restrict__ x, float* __restrict__ out, int total) {
    int i = blockIdx.x * blockDim.x + threadIdx.x;   // over B*T*256
    if (i >= total) return;
    int c = i & 255, bt = i >> 8;
    const _Float16* p = x + ((size_t)bt * NJOINT) * 256 + c;
    float s = 0.0f;
    for (int n = 0; n < NJOINT; ++n) s += (float)p[(size_t)n * 256];
    out[i] = s * (1.0f / (float)NJOINT);
}

// attention pooling per batch element: softmax_T( w2 . tanh(W1 h + b1) + b2 )
__global__ __launch_bounds__(256) void k_attn_pool(
    const float* __restrict__ hm,   // [B,T,256]
    const float* __restrict__ w1, const float* __restrict__ b1,   // [64,256],[64]
    const float* __restrict__ w2, const float* __restrict__ b2,   // [64],[1]
    float* __restrict__ pooled)     // [B,256]
{
    const int b = blockIdx.x, tid = threadIdx.x;
    __shared__ float sm_tmp[64];
    __shared__ float sm_a[TLEN];
    __shared__ float sm_w[TLEN];

    for (int t = 0; t < TLEN; ++t) {
        const float* h = hm + ((size_t)b * TLEN + t) * 256;
        if (tid < 64) {
            float s = b1[tid];
            const float* wr = w1 + tid * 256;
            for (int k = 0; k < 256; ++k) s = fmaf(wr[k], h[k], s);
            sm_tmp[tid] = tanhf(s);
        }
        __syncthreads();
        if (tid == 0) {
            float s = b2[0];
            for (int k = 0; k < 64; ++k) s = fmaf(w2[k], sm_tmp[k], s);
            sm_a[t] = s;
        }
        __syncthreads();
    }
    if (tid == 0) {
        float mx = -1e30f;
        for (int t = 0; t < TLEN; ++t) mx = fmaxf(mx, sm_a[t]);
        float den = 0.0f;
        for (int t = 0; t < TLEN; ++t) den += expf(sm_a[t] - mx);
        float inv = 1.0f / den;
        for (int t = 0; t < TLEN; ++t) sm_w[t] = expf(sm_a[t] - mx) * inv;
    }
    __syncthreads();
    {
        float s = 0.0f;
        const float* hb = hm + (size_t)b * TLEN * 256 + tid;
        for (int t = 0; t < TLEN; ++t) s = fmaf(sm_w[t], hb[(size_t)t * 256], s);
        pooled[(size_t)b * 256 + tid] = s;
    }
}

// LayerNorm + cls1(gelu) + cls2 -> logits [B,100]
__global__ __launch_bounds__(256) void k_head(
    const float* __restrict__ pooled,
    const float* __restrict__ lng, const float* __restrict__ lnb,
    const float* __restrict__ w1, const float* __restrict__ b1,   // [256,256],[256]
    const float* __restrict__ w2, const float* __restrict__ b2,   // [100,256],[100]
    float* __restrict__ out)
{
    const int b = blockIdx.x, tid = threadIdx.x;
    __shared__ float red[256];
    __shared__ float z[256];
    __shared__ float z1[256];

    float v = pooled[(size_t)b * 256 + tid];
    red[tid] = v; __syncthreads();
    for (int s = 128; s > 0; s >>= 1) { if (tid < s) red[tid] += red[tid + s]; __syncthreads(); }
    float mu = red[0] * (1.0f / 256.0f);
    __syncthreads();
    float dv = v - mu;
    red[tid] = dv * dv; __syncthreads();
    for (int s = 128; s > 0; s >>= 1) { if (tid < s) red[tid] += red[tid + s]; __syncthreads(); }
    float var = red[0] * (1.0f / 256.0f);
    z[tid] = dv * rsqrtf(var + 1e-5f) * lng[tid] + lnb[tid];
    __syncthreads();

    float s1 = b1[tid];
    const float* wr = w1 + tid * 256;
    for (int k = 0; k < 256; ++k) s1 = fmaf(wr[k], z[k], s1);
    z1[tid] = gelu_exact(s1);
    __syncthreads();

    if (tid < NCLS) {
        float s2 = b2[tid];
        const float* wr2 = w2 + tid * 256;
        for (int k = 0; k < 256; ++k) s2 = fmaf(wr2[k], z1[k], s2);
        out[(size_t)b * NCLS + tid] = s2;
    }
}

// ---------------------------------------------------------------------------
static inline int cdiv(long long a, long long b) { return (int)((a + b - 1) / b); }

extern "C" void kernel_launch(void* const* d_in, const int* in_sizes, int n_in,
                              void* d_out, int out_size, void* d_ws, size_t ws_size,
                              hipStream_t stream) {
    // ---- inputs, flattened in setup_inputs() dict insertion order ----
    const float* x        = (const float*)d_in[0];   // [B,T,N,6]
    const int*   s_ei     = (const int*)d_in[1];     // [2,Es]
    const int*   t_ei     = (const int*)d_in[2];     // [2,Et]
    const float* ip_w     = (const float*)d_in[3];   // [64,6]
    const float* ip_b     = (const float*)d_in[4];
    const float* sp1_w    = (const float*)d_in[5];   // [128,64]
    const float* sp1_b    = (const float*)d_in[6];
    const float* sp1_bng  = (const float*)d_in[7];
    const float* sp1_bnb  = (const float*)d_in[8];
    const float* sp1_rw   = (const float*)d_in[9];   // [128,64]
    const float* sp1_rb   = (const float*)d_in[10];
    const float* sp2_w    = (const float*)d_in[11];  // [128,128]
    const float* sp2_b    = (const float*)d_in[12];
    const float* sp2_bng  = (const float*)d_in[13];
    const float* sp2_bnb  = (const float*)d_in[14];
    const float* tp1_w    = (const float*)d_in[15];  // [256,128]
    const float* tp1_b    = (const float*)d_in[16];
    const float* tp1_bng  = (const float*)d_in[17];
    const float* tp1_bnb  = (const float*)d_in[18];
    const float* tp1_rw   = (const float*)d_in[19];  // [256,128]
    const float* tp1_rb   = (const float*)d_in[20];
    const float* tp2_w    = (const float*)d_in[21];  // [256,256]
    const float* tp2_b    = (const float*)d_in[22];
    const float* tp2_bng  = (const float*)d_in[23];
    const float* tp2_bnb  = (const float*)d_in[24];
    const float* at1_w    = (const float*)d_in[25];  // [64,256]
    const float* at1_b    = (const float*)d_in[26];
    const float* at2_w    = (const float*)d_in[27];  // [1,64]
    const float* at2_b    = (const float*)d_in[28];
    const float* ln_g     = (const float*)d_in[29];
    const float* ln_b     = (const float*)d_in[30];
    const float* c1_w     = (const float*)d_in[31];  // [256,256]
    const float* c1_b     = (const float*)d_in[32];
    const float* c2_w     = (const float*)d_in[33];  // [100,256]
    const float* c2_b     = (const float*)d_in[34];
    const float* sp1_rm   = (const float*)d_in[35];
    const float* sp1_rv   = (const float*)d_in[36];
    const float* sp2_rm   = (const float*)d_in[37];
    const float* sp2_rv   = (const float*)d_in[38];
    const float* tp1_rm   = (const float*)d_in[39];
    const float* tp1_rv   = (const float*)d_in[40];
    const float* tp2_rm   = (const float*)d_in[41];
    const float* tp2_rv   = (const float*)d_in[42];
    (void)n_in; (void)out_size; (void)ws_size;

    const int M  = NNODES;
    const int Es = in_sizes[1] / 2;
    const int Et = in_sizes[2] / 2;
    const int* s_src = s_ei;          const int* s_dst = s_ei + Es;
    const int* t_src = t_ei;          const int* t_dst = t_ei + Et;

    // ---- workspace layout ----
    char* ws = (char*)d_ws;
    _Float16* buf_xh = (_Float16*)ws;  ws += (size_t)M * 256 * sizeof(_Float16);
    float* buf_h  = (float*)ws;        ws += (size_t)M * 256 * sizeof(float);
    float* buf_r  = (float*)ws;        ws += (size_t)M * 256 * sizeof(float);
    float* buf_a  = (float*)ws;        ws += (size_t)M * 256 * sizeof(float);
    float* dinv_s = (float*)ws;        ws += (size_t)M * sizeof(float);
    float* dinv_t = (float*)ws;        ws += (size_t)M * sizeof(float);
    float* hmean  = (float*)ws;        ws += (size_t)NBATCH * TLEN * 256 * sizeof(float);
    float* pooled = (float*)ws;        ws += (size_t)NBATCH * 256 * sizeof(float);
    _Float16* wh_sp1  = (_Float16*)ws; ws += (size_t)128 * 64  * sizeof(_Float16);
    _Float16* wh_sp1r = (_Float16*)ws; ws += (size_t)128 * 64  * sizeof(_Float16);
    _Float16* wh_sp2  = (_Float16*)ws; ws += (size_t)128 * 128 * sizeof(_Float16);
    _Float16* wh_tp1  = (_Float16*)ws; ws += (size_t)256 * 128 * sizeof(_Float16);
    _Float16* wh_tp1r = (_Float16*)ws; ws += (size_t)256 * 128 * sizeof(_Float16);
    _Float16* wh_tp2  = (_Float16*)ws; ws += (size_t)256 * 256 * sizeof(_Float16);

    const int TB = 256;
    auto gemm = [&](const _Float16* X, const _Float16* W, const float* bias,
                    float* out, int N, int K) {
        dim3 grid(cdiv(M / 16, 8), N / 64);
        k_wmma_gemm<<<grid, 256, 0, stream>>>(X, W, bias, out, M, N, K);
    };

    // ---- f16 weight conversion (tiny, once per launch) ----
    k_cvt_f32_f16<<<cdiv(128 * 64,  TB), TB, 0, stream>>>(sp1_w,  wh_sp1,  128 * 64);
    k_cvt_f32_f16<<<cdiv(128 * 64,  TB), TB, 0, stream>>>(sp1_rw, wh_sp1r, 128 * 64);
    k_cvt_f32_f16<<<cdiv(128 * 128, TB), TB, 0, stream>>>(sp2_w,  wh_sp2,  128 * 128);
    k_cvt_f32_f16<<<cdiv(256 * 128, TB), TB, 0, stream>>>(tp1_w,  wh_tp1,  256 * 128);
    k_cvt_f32_f16<<<cdiv(256 * 128, TB), TB, 0, stream>>>(tp1_rw, wh_tp1r, 256 * 128);
    k_cvt_f32_f16<<<cdiv(256 * 256, TB), TB, 0, stream>>>(tp2_w,  wh_tp2,  256 * 256);

    // ---- degree norms (self-loop: deg starts at 1) ----
    k_fill<<<cdiv(M, TB), TB, 0, stream>>>(dinv_s, M, 1.0f);
    k_fill<<<cdiv(M, TB), TB, 0, stream>>>(dinv_t, M, 1.0f);
    k_deg_count<<<cdiv(Es, TB), TB, 0, stream>>>(s_dst, dinv_s, Es);
    k_deg_count<<<cdiv(Et, TB), TB, 0, stream>>>(t_dst, dinv_t, Et);
    k_rsqrt_inplace<<<cdiv(M, TB), TB, 0, stream>>>(dinv_s, M);
    k_rsqrt_inplace<<<cdiv(M, TB), TB, 0, stream>>>(dinv_t, M);

    // ---- input projection -> buf_xh [M,64] f16 ----
    k_inproj<<<cdiv((long long)M * 64, TB), TB, 0, stream>>>(x, ip_w, ip_b, buf_xh, M);

    // ---- sp1: 64 -> 128 (spatial edges, linear residual) ----
    gemm(buf_xh, wh_sp1,  nullptr, buf_h, 128, 64);
    gemm(buf_xh, wh_sp1r, sp1_rb,  buf_r, 128, 64);
    k_agg_init<<<cdiv((long long)M * 128, TB), TB, 0, stream>>>(buf_h, dinv_s, sp1_b, buf_a, M * 128, 128);
    k_agg_edges<<<cdiv((long long)Es * 32, TB), TB, 0, stream>>>(buf_h, s_src, s_dst, dinv_s, buf_a, Es, 128);
    k_bn_resf_gelu<<<cdiv((long long)M * 128, TB), TB, 0, stream>>>(buf_a, sp1_rm, sp1_rv, sp1_bng, sp1_bnb, buf_r, buf_xh, M * 128, 128);

    // ---- sp2: 128 -> 128 (identity residual) ----
    gemm(buf_xh, wh_sp2, nullptr, buf_h, 128, 128);
    k_agg_init<<<cdiv((long long)M * 128, TB), TB, 0, stream>>>(buf_h, dinv_s, sp2_b, buf_a, M * 128, 128);
    k_agg_edges<<<cdiv((long long)Es * 32, TB), TB, 0, stream>>>(buf_h, s_src, s_dst, dinv_s, buf_a, Es, 128);
    k_bn_resh_gelu<<<cdiv((long long)M * 128, TB), TB, 0, stream>>>(buf_a, sp2_rm, sp2_rv, sp2_bng, sp2_bnb, buf_xh, buf_xh, M * 128, 128);

    // ---- tp1: 128 -> 256 (temporal edges, linear residual) ----
    gemm(buf_xh, wh_tp1,  nullptr, buf_h, 256, 128);
    gemm(buf_xh, wh_tp1r, tp1_rb,  buf_r, 256, 128);
    k_agg_init<<<cdiv((long long)M * 256, TB), TB, 0, stream>>>(buf_h, dinv_t, tp1_b, buf_a, M * 256, 256);
    k_agg_edges<<<cdiv((long long)Et * 64, TB), TB, 0, stream>>>(buf_h, t_src, t_dst, dinv_t, buf_a, Et, 256);
    k_bn_resf_gelu<<<cdiv((long long)M * 256, TB), TB, 0, stream>>>(buf_a, tp1_rm, tp1_rv, tp1_bng, tp1_bnb, buf_r, buf_xh, M * 256, 256);

    // ---- tp2: 256 -> 256 (identity residual) ----
    gemm(buf_xh, wh_tp2, nullptr, buf_h, 256, 256);
    k_agg_init<<<cdiv((long long)M * 256, TB), TB, 0, stream>>>(buf_h, dinv_t, tp2_b, buf_a, M * 256, 256);
    k_agg_edges<<<cdiv((long long)Et * 64, TB), TB, 0, stream>>>(buf_h, t_src, t_dst, dinv_t, buf_a, Et, 256);
    k_bn_resh_gelu<<<cdiv((long long)M * 256, TB), TB, 0, stream>>>(buf_a, tp2_rm, tp2_rv, tp2_bng, tp2_bnb, buf_xh, buf_xh, M * 256, 256);

    // ---- mean over joints, attention pooling, head ----
    k_mean_joints<<<cdiv(NBATCH * TLEN * 256, TB), TB, 0, stream>>>(buf_xh, hmean, NBATCH * TLEN * 256);
    k_attn_pool<<<NBATCH, 256, 0, stream>>>(hmean, at1_w, at1_b, at2_w, at2_b, pooled);
    k_head<<<NBATCH, 256, 0, stream>>>(pooled, ln_g, ln_b, c1_w, c1_b, c2_w, c2_b, (float*)d_out);
}
